// NoisyTopKRouter_64106681860775
// MI455X (gfx1250) — matrix operand order repaired and checked by reading, not compiled, not run on hardware
//
#include <hip/hip_runtime.h>
#include <hip/hip_bf16.h>
#include <stdint.h>

// Problem constants (from the reference): B=4, S=4096, D=2048, E=64, K=2
#define TOK   16384          // B*S tokens
#define DDIM  2048
#define EEXP  64
#define KC    128            // K-chunk staged per iteration (4 WMMA k-steps)
#define NOISE_EPS 0.2f

typedef __attribute__((ext_vector_type(16))) __bf16 v16bf;
typedef __attribute__((ext_vector_type(8)))  float  v8f;

// LDS layout (bytes):
//   [0      , 32768) : bf16 weight fragments, 32 frag-ids x 32 lanes x 32B
//                      (reused as per-wave epilogue scratch: wave w -> w*8192)
//   [32768  , 49152) : eps tile for this block, 64 tokens x 64 experts f32
//   [49152  , 49664) : Psum[64] + cnt[64] partial reductions
#define SMEM_BYTES   49664
#define FRAG_DWORDS  8192
#define EPS_DWORD    8192
#define PC_DWORD     12288

__device__ __forceinline__ uint32_t pack_bf16x2(float lo, float hi) {
  __bf16 a = (__bf16)lo, b = (__bf16)hi;
  uint16_t ua, ub;
  __builtin_memcpy(&ua, &a, 2);
  __builtin_memcpy(&ub, &b, 2);
  return (uint32_t)ua | ((uint32_t)ub << 16);
}

__device__ __forceinline__ v16bf load_b_frag(const uint32_t* p) {
  union { int4 i4[2]; v16bf v; } u;
  u.i4[0] = ((const int4*)p)[0];
  u.i4[1] = ((const int4*)p)[1];
  return u.v;
}

// A fragment, contiguous k order used here, matching the B staging:
// lane<16: row=lane,  K = kbase+0..15 (v0..7)
// lane>=16: row=lane-16, same but kbase+=16
__device__ __forceinline__ v16bf load_a_frag(const float* xr) {
  float4 p0 = ((const float4*)xr)[0];
  float4 p1 = ((const float4*)xr)[1];
  float4 q0 = ((const float4*)(xr + 8))[0];
  float4 q1 = ((const float4*)(xr + 8))[1];
  v16bf a;
  a[0]=(__bf16)p0.x; a[1]=(__bf16)p0.y; a[2]=(__bf16)p0.z; a[3]=(__bf16)p0.w;
  a[4]=(__bf16)p1.x; a[5]=(__bf16)p1.y; a[6]=(__bf16)p1.z; a[7]=(__bf16)p1.w;
  a[8]=(__bf16)q0.x; a[9]=(__bf16)q0.y; a[10]=(__bf16)q0.z; a[11]=(__bf16)q0.w;
  a[12]=(__bf16)q1.x; a[13]=(__bf16)q1.y; a[14]=(__bf16)q1.z; a[15]=(__bf16)q1.w;
  return a;
}

__device__ __forceinline__ v8f wmma_bf16(v16bf a, v16bf b, v8f c) {
  return __builtin_amdgcn_wmma_f32_16x16x32_bf16(false, a, false, b,
                                                 (short)0, c, false, false);
}

__global__ void __launch_bounds__(128)
router_wmma_kernel(const float* __restrict__ x,
                   const float* __restrict__ w_gate,
                   const float* __restrict__ w_noise,
                   const float* __restrict__ eps,
                   float* __restrict__ gates,
                   float* __restrict__ ws) {
  __shared__ __attribute__((aligned(16))) unsigned char smem[SMEM_BYTES];
  uint32_t* fragLds = (uint32_t*)smem;
  float*    epsLds  = (float*)smem + EPS_DWORD;
  float*    pcLds   = (float*)smem + PC_DWORD;   // [0..63]=Psum, [64..127]=cnt

  const int tid  = threadIdx.x;
  const int wv   = tid >> 5;
  const int lane = tid & 31;
  const int hi   = lane >> 4;
  const int col0 = lane & 15;
  const int blk  = blockIdx.x;
  const int tokBlk = blk * 64 + wv * 16;          // this wave's 16-token tile
  const int row    = tokBlk + col0;               // x row this lane loads

  // zero per-block reduction scratch (visible after first __syncthreads)
  pcLds[tid] = 0.0f;

  // ---- async prefetch of eps tile (64 tokens x 64 experts = 16KB) ----
  {
    unsigned long long sbase =
        (unsigned long long)(uintptr_t)(eps + (size_t)blk * 4096);
#pragma unroll
    for (int it = 0; it < 8; ++it) {
      unsigned voff   = (unsigned)(tid * 16 + it * 2048);
      unsigned ldsOff = (unsigned)(uintptr_t)((unsigned char*)epsLds + voff);
      asm volatile("global_load_async_to_lds_b128 %0, %1, %2 offset:0"
                   :: "v"(ldsOff), "v"(voff), "s"(sbase) : "memory");
    }
  }

  v8f accG[4] = {};   // clean logits, 4 N-tiles of 16x16 f32
  v8f accN[4] = {};   // raw noise logits

  for (int kc = 0; kc < DDIM; kc += KC) {
    __syncthreads();   // previous chunk fully consumed
    // ---- stage bf16 weight fragments for this K-chunk (both matrices) ----
    // B-fragment layout (32x16 bf16): lanes 0-15 hold K=0..15 (v = K/2),
    // lanes 16-31 hold K=16..31; packed pairs (Keven lo, Kodd hi).
    for (int j = tid; j < FRAG_DWORDS; j += 128) {
      int m  = j >> 12;        // 0 = gate, 1 = noise
      int r  = j & 4095;
      int kp = r >> 6;         // K-pair index within chunk, 0..63
      int e  = r & 63;
      const float* wm = m ? w_noise : w_gate;
      float f0 = wm[(kc + 2 * kp)     * EEXP + e];
      float f1 = wm[(kc + 2 * kp + 1) * EEXP + e];
      int ks   = kp >> 4;      // which 32-K WMMA step, 0..3
      int kpl  = kp & 15;
      int ln   = (e & 15) + ((kpl >> 3) << 4);
      int v    = kpl & 7;
      int t    = e >> 4;
      fragLds[((((m * 4 + ks) * 4 + t) * 32 + ln) << 3) + v] = pack_bf16x2(f0, f1);
    }
    __syncthreads();

#pragma unroll
    for (int ks = 0; ks < 4; ++ks) {
      v16bf a = load_a_frag(x + (size_t)row * DDIM + kc + ks * 32 + hi * 16);
#pragma unroll
      for (int t = 0; t < 4; ++t) {
        v16bf bg = load_b_frag(fragLds + ((((0 * 4 + ks) * 4 + t) * 32 + lane) << 3));
        accG[t] = wmma_bf16(a, bg, accG[t]);
      }
#pragma unroll
      for (int t = 0; t < 4; ++t) {
        v16bf bn = load_b_frag(fragLds + ((((1 * 4 + ks) * 4 + t) * 32 + lane) << 3));
        accN[t] = wmma_bf16(a, bn, accN[t]);
      }
    }
  }

  // ---- epilogue ----
  asm volatile("s_wait_asynccnt 0x0" ::: "memory");
  __syncthreads();   // eps ready from all waves; frag LDS free for reuse

  float* cleanS = (float*)(smem + wv * 8192);   // [16][64]
  float* noisyS = cleanS + 1024;                // [16][64]

  // C/D layout: tile t, vgpr r, lane -> (M = r + 8*hi, e = col0 + 16*t)
#pragma unroll
  for (int t = 0; t < 4; ++t) {
    int e = col0 + 16 * t;
#pragma unroll
    for (int r = 0; r < 8; ++r) {
      int M = r + 8 * hi;
      float cl = accG[t][r];
      float rw = accN[t][r];
      float sp = (rw > 20.0f) ? rw : log1pf(__expf(rw));   // softplus
      float ns = cl + (sp + NOISE_EPS) * epsLds[(wv * 16 + M) * 64 + e];
      cleanS[M * 64 + e] = cl;
      noisyS[M * 64 + e] = ns;
    }
  }
  // same-wave LDS ops are in order: safe to read back without a barrier

  const int tk    = col0;        // token within wave tile (2 lanes/token)
  const int half  = hi;          // this lane scans experts [32*half, 32*half+32)
  const int ebase = half * 32;

  float v0 = -3.4e38f, v1 = -3.4e38f; int i0 = 0, i1 = 0;
  float cmax = -3.4e38f;
  for (int j = 0; j < 32; ++j) {
    int e = ebase + j;
    float nv = noisyS[tk * 64 + e];
    if (nv > v0)      { v1 = v0; i1 = i0; v0 = nv; i0 = e; }
    else if (nv > v1) { v1 = nv; i1 = e; }
    cmax = fmaxf(cmax, cleanS[tk * 64 + e]);
  }
  // merge the two halves of each token
  float pv0 = __shfl_xor(v0, 16, 32); int pi0 = __shfl_xor(i0, 16, 32);
  float pv1 = __shfl_xor(v1, 16, 32); int pi1 = __shfl_xor(i1, 16, 32);
  if (pv0 > v0) {
    float nv1 = (v0 > pv1) ? v0 : pv1;
    int   ni1 = (v0 > pv1) ? i0 : pi1;
    v0 = pv0; i0 = pi0; v1 = nv1; i1 = ni1;
  } else if (pv0 > v1) { v1 = pv0; i1 = pi0; }
  cmax = fmaxf(cmax, __shfl_xor(cmax, 16, 32));

  float es = 0.0f;
  for (int j = 0; j < 32; ++j)
    es += __expf(cleanS[tk * 64 + ebase + j] - cmax);
  es += __shfl_xor(es, 16, 32);
  float rden = 1.0f / es;

  // softmax over the 2 selected logits (v0 >= v1)
  float g0 = 1.0f / (1.0f + __expf(v1 - v0));
  float g1 = 1.0f - g0;

  // dense gates write (each lane owns 32 experts of its token)
  float* gp = gates + (size_t)(tokBlk + tk) * EEXP;
#pragma unroll
  for (int j = 0; j < 8; ++j) {
    int e0 = ebase + 4 * j;
    float4 gv;
    gv.x = (e0 + 0 == i0) ? g0 : (e0 + 0 == i1) ? g1 : 0.0f;
    gv.y = (e0 + 1 == i0) ? g0 : (e0 + 1 == i1) ? g1 : 0.0f;
    gv.z = (e0 + 2 == i0) ? g0 : (e0 + 2 == i1) ? g1 : 0.0f;
    gv.w = (e0 + 3 == i0) ? g0 : (e0 + 3 == i1) ? g1 : 0.0f;
    *(float4*)(gp + e0) = gv;
  }

  // aux-loss partials: P (softmax of clean) and f (top-2 counts)
  for (int j = 0; j < 32; ++j) {
    int e = ebase + j;
    atomicAdd(&pcLds[e], __expf(cleanS[tk * 64 + e] - cmax) * rden);
  }
  if (half == 0) {
    atomicAdd(&pcLds[64 + i0], 1.0f);
    atomicAdd(&pcLds[64 + i1], 1.0f);
  }
  __syncthreads();
  atomicAdd(&ws[tid], pcLds[tid]);
}

__global__ void router_zero_ws(float* ws) { ws[threadIdx.x] = 0.0f; }

__global__ void router_finalize(const float* __restrict__ ws,
                                float* __restrict__ out_aux) {
  int l = threadIdx.x;   // 32 threads
  float s = 0.0f;
  for (int e = l; e < 64; e += 32) s += ws[e] * ws[64 + e];
#pragma unroll
  for (int off = 16; off > 0; off >>= 1) s += __shfl_xor(s, off, 32);
  if (l == 0)
    out_aux[0] = (float)EEXP * s / ((float)TOK * (float)TOK);
}

extern "C" void kernel_launch(void* const* d_in, const int* in_sizes, int n_in,
                              void* d_out, int out_size, void* d_ws, size_t ws_size,
                              hipStream_t stream) {
  const float* x       = (const float*)d_in[0];
  const float* w_gate  = (const float*)d_in[1];
  const float* w_noise = (const float*)d_in[2];
  const float* eps     = (const float*)d_in[3];
  float* gates = (float*)d_out;                 // [16384 x 64]
  float* aux   = (float*)d_out + (size_t)TOK * EEXP;
  float* ws    = (float*)d_ws;                  // 128 floats: Psum + cnt

  router_zero_ws<<<1, 128, 0, stream>>>(ws);
  router_wmma_kernel<<<TOK / 64, 128, 0, stream>>>(x, w_gate, w_noise, eps,
                                                   gates, ws);
  router_finalize<<<1, 32, 0, stream>>>(ws, aux);
}
